// BeyazKusAIMoE_36515811951170
// MI455X (gfx1250) — matrix operand order repaired and checked by reading, not compiled, not run on hardware
//
#include <hip/hip_runtime.h>

// ---------------------------------------------------------------------------
// MoE (DeepSeek-style) for MI455X / gfx1250, wave32 + WMMA bf16.
//
// Shapes: T=2048 tokens, D=1024, E=64 routed experts (top-8), H=256,
//         ES=2 shared experts, HS=2048.
// Strategy: one streaming pass converts fp32 weights to bf16 *transposed*
// (K-contiguous) so each v_wmma_f32_16x16x32_bf16 B-fragment is 2x16B loads
// per lane. Routed experts are computed *gathered* (only top-k pairs): 8x
// less FLOPs than the dense reference formulation, same result since cw=0
// for unselected experts. Async LDS loads (global_load_async_to_lds_b128 +
// s_wait_asynccnt) stage X tiles. Routed contributions use
// global_atomic_add_f32 on top of the densely-written shared-expert output.
// ---------------------------------------------------------------------------

#define T_TOK 2048
#define DIM   1024
#define NEXP  64
#define TOPK  8
#define HMOE  256
#define NSH   2
#define HSH   2048

typedef __bf16 bf16_t;
typedef __attribute__((ext_vector_type(16))) __bf16 v16bf;
typedef __attribute__((ext_vector_type(8)))  float  v8f;

#define USE_ASYNC_LDS 1

// ---- WMMA fragment helpers (layouts per CDNA5 ISA 7.12.2) ------------------

// A-matrix 16x32 bf16, row-major source `row` (one matrix row per M):
// lane<16 holds M=lane, K={k0..k0+7, k0+16..k0+23}; lane>=16 shifts K by 8.
__device__ __forceinline__ v16bf frag_a(const bf16_t* row, int k0, int lh) {
  v16bf r;
  uint4* p = reinterpret_cast<uint4*>(&r);
  p[0] = *reinterpret_cast<const uint4*>(row + k0 + lh * 8);
  p[1] = *reinterpret_cast<const uint4*>(row + k0 + 16 + lh * 8);
  return r;
}

// B-matrix 32x16 bf16 from transposed weights (row = one N column,
// K-contiguous): lane<16 holds K={k0..k0+7, k0+16..k0+23} of column lane;
// lane>=16 shifts K by 8 (column lane-16).
__device__ __forceinline__ v16bf frag_b(const bf16_t* colrow, int k0, int lh) {
  v16bf r;
  uint4* p = reinterpret_cast<uint4*>(&r);
  const bf16_t* s = colrow + k0 + lh * 8;
  p[0] = *reinterpret_cast<const uint4*>(s);
  p[1] = *reinterpret_cast<const uint4*>(s + 16);
  return r;
}

__device__ __forceinline__ v8f wmma_bf16(v16bf a, v16bf b, v8f c) {
  // (neg_a, A, neg_b, B, c_mod, C, reuse_a, reuse_b)
  return __builtin_amdgcn_wmma_f32_16x16x32_bf16(false, a, false, b,
                                                 (short)0, c, false, false);
}

// ---- Async global->LDS staging (CDNA5 ASYNCcnt path) -----------------------

__device__ __forceinline__ void async_copy16(const void* g, void* lds) {
#if USE_ASYNC_LDS
  // Low 32 bits of a generic LDS pointer are the wave-relative LDS address.
  unsigned laddr = (unsigned)(unsigned long long)lds;
  asm volatile("global_load_async_to_lds_b128 %0, %1, off"
               :: "v"(laddr), "v"(g) : "memory");
#else
  *reinterpret_cast<uint4*>(lds) = *reinterpret_cast<const uint4*>(g);
#endif
}

__device__ __forceinline__ void async_wait() {
#if USE_ASYNC_LDS
  asm volatile("s_wait_asynccnt 0" ::: "memory");
#endif
}

__device__ __forceinline__ float silu(float v) {
  return v * (1.f / (1.f + __expf(-v)));
}

// ---- Precision conversion kernels ------------------------------------------

__global__ void cvt_bf16_kernel(const float* __restrict__ in,
                                bf16_t* __restrict__ out, long n) {
  long i  = (long)blockIdx.x * blockDim.x + threadIdx.x;
  long st = (long)gridDim.x * blockDim.x;
  for (; i < n; i += st) out[i] = (bf16_t)in[i];
}

// in [Bt][Kd][Nd] fp32  ->  out [Bt][Nd][Kd] bf16 (K-contiguous for WMMA B)
__global__ void transpose_cvt_kernel(const float* __restrict__ in,
                                     bf16_t* __restrict__ out,
                                     int Bt, int Kd, int Nd) {
  long n  = (long)Bt * Kd * Nd;
  long i  = (long)blockIdx.x * blockDim.x + threadIdx.x;
  long st = (long)gridDim.x * blockDim.x;
  long plane = (long)Kd * Nd;
  for (; i < n; i += st) {
    long b = i / plane, r = i - b * plane;
    long nn = r / Kd, kk = r - nn * Kd;
    out[i] = (bf16_t)in[(b * Kd + kk) * Nd + nn];
  }
}

// ---- Router: softmax + top-8 + renormalize + expert token lists ------------

__global__ __launch_bounds__(64)
void router_kernel(const float* __restrict__ x, const float* __restrict__ rw,
                   const float* __restrict__ rb, float* __restrict__ cw,
                   int* __restrict__ counts, int* __restrict__ lists) {
  __shared__ float xrow[DIM];
  __shared__ float q[NEXP];
  const int t = blockIdx.x, tid = threadIdx.x;
  for (int d = tid; d < DIM; d += 64) xrow[d] = x[(size_t)t * DIM + d];
  __syncthreads();
  float acc = rb[tid];
  for (int d = 0; d < DIM; ++d) acc += xrow[d] * rw[(size_t)d * NEXP + tid];
  q[tid] = acc;
  cw[(size_t)t * NEXP + tid] = 0.f;
  __syncthreads();
  if (tid == 0) {
    float m = q[0];
    for (int e = 1; e < NEXP; ++e) m = fmaxf(m, q[e]);
    float p[NEXP];
    for (int e = 0; e < NEXP; ++e) p[e] = __expf(q[e] - m);
    // top-k of softmax == top-k of exp(l-m); renormalization cancels the
    // softmax denominator, so weights are exp terms over their top-k sum.
    int sel[TOPK]; float swt[TOPK]; float s = 0.f;
    for (int k = 0; k < TOPK; ++k) {
      int bi = 0; float bv = -1.f;
      for (int e = 0; e < NEXP; ++e) if (p[e] > bv) { bv = p[e]; bi = e; }
      sel[k] = bi; swt[k] = bv; s += bv; p[bi] = -2.f;
    }
    float inv = 1.f / s;
    for (int k = 0; k < TOPK; ++k) {
      int e = sel[k];
      cw[(size_t)t * NEXP + e] = swt[k] * inv;
      int idx = atomicAdd(&counts[e], 1);
      lists[(size_t)e * T_TOK + idx] = t;
    }
  }
}

// ---- Shared experts, GEMM1: Hs[es][t][HS] = silu(x @ sw1 + sb1) ------------
// Block = 256 thr (8 waves), tile M=16 tokens x N=128 cols, K=D=1024.

__global__ __launch_bounds__(256)
void shared_h_kernel(const bf16_t* __restrict__ xbf,
                     const bf16_t* __restrict__ sw1T,
                     const float* __restrict__ sb1, bf16_t* __restrict__ Hs) {
  __shared__ __align__(16) bf16_t Xs[16][DIM];
  const int mt = blockIdx.x, nb = blockIdx.y, es = blockIdx.z;
  const int tid = threadIdx.x;
  {
    int row = tid >> 4, seg = tid & 15;
    const bf16_t* src = xbf + (size_t)(mt * 16 + row) * DIM;
    for (int j = 0; j < 8; ++j) {
      int el = (seg * 8 + j) * 8;
      async_copy16(src + el, &Xs[row][el]);
    }
    async_wait();
  }
  __syncthreads();
  const int wid = tid >> 5, lane = tid & 31, lh = lane >> 4, rowA = lane & 15;
  const int colg = nb * 128 + wid * 16 + (lane & 15);
  const bf16_t* brow = sw1T + ((size_t)es * HSH + colg) * DIM;
  v8f acc = {};
  for (int k0 = 0; k0 < DIM; k0 += 32)
    acc = wmma_bf16(frag_a(&Xs[rowA][0], k0, lh), frag_b(brow, k0, lh), acc);
  float bb = sb1[es * HSH + colg];
  for (int i = 0; i < 8; ++i) {
    int row = lh * 8 + i;
    float v = silu(acc[i] + bb);
    Hs[((size_t)es * T_TOK + (mt * 16 + row)) * HSH + colg] = (bf16_t)v;
  }
}

// ---- Shared experts, GEMM2: out = sum_es Hs_es @ sw2_es + sum_es sb2 -------
// Dense write: initializes every element of out.

__global__ __launch_bounds__(256)
void shared_out_kernel(const bf16_t* __restrict__ Hs,
                       const bf16_t* __restrict__ sw2T,
                       const float* __restrict__ sb2, float* __restrict__ out) {
  const int mt = blockIdx.x, nb = blockIdx.y;
  const int tid = threadIdx.x, wid = tid >> 5, lane = tid & 31;
  const int lh = lane >> 4, rowA = lane & 15;
  const int colg = nb * 128 + wid * 16 + (lane & 15);
  v8f acc = {};
  for (int es = 0; es < NSH; ++es) {
    const bf16_t* arow = Hs + ((size_t)es * T_TOK + mt * 16 + rowA) * HSH;
    const bf16_t* brow = sw2T + ((size_t)es * DIM + colg) * HSH;
    for (int k0 = 0; k0 < HSH; k0 += 32)
      acc = wmma_bf16(frag_a(arow, k0, lh), frag_b(brow, k0, lh), acc);
  }
  float bb = sb2[colg] + sb2[DIM + colg];
  for (int i = 0; i < 8; ++i) {
    int row = lh * 8 + i;
    out[(size_t)(mt * 16 + row) * DIM + colg] = acc[i] + bb;
  }
}

// ---- Routed experts, fused gathered MLP ------------------------------------
// Grid (tile, expert); block processes 16 gathered tokens of one expert:
// GEMM1 (K=1024 -> H=256) + silu into LDS, GEMM2 (K=256 -> D=1024),
// scale by combine weight, atomic-add into out.

__global__ __launch_bounds__(256)
void routed_kernel(const bf16_t* __restrict__ xbf, const bf16_t* __restrict__ w1T,
                   const bf16_t* __restrict__ w2T, const float* __restrict__ b1,
                   const float* __restrict__ b2, const float* __restrict__ cw,
                   const int* __restrict__ counts, const int* __restrict__ lists,
                   float* __restrict__ out) {
  const int e = blockIdx.y, tile = blockIdx.x, tid = threadIdx.x;
  const int n = counts[e];
  if (tile * 16 >= n) return;  // uniform per block

  __shared__ __align__(16) bf16_t Xs[16][DIM];
  __shared__ __align__(16) bf16_t Hsh[16][HMOE];
  __shared__ int   toks[16];
  __shared__ float cwv[16];

  if (tid < 16) {
    int idx = tile * 16 + tid;
    bool ok = idx < n;
    int t = ok ? lists[(size_t)e * T_TOK + idx] : 0;
    toks[tid] = t;
    cwv[tid]  = ok ? cw[(size_t)t * NEXP + e] : 0.f;  // pad rows contribute 0
  }
  __syncthreads();
  {
    int row = tid >> 4, seg = tid & 15;
    const bf16_t* src = xbf + (size_t)toks[row] * DIM;
    for (int j = 0; j < 8; ++j) {
      int el = (seg * 8 + j) * 8;
      async_copy16(src + el, &Xs[row][el]);
    }
    async_wait();
  }
  __syncthreads();

  const int wid = tid >> 5, lane = tid & 31, lh = lane >> 4, rowA = lane & 15;

  // GEMM1: H = silu(X @ W1 + b1); 16 N-tiles over 8 waves (2 each).
  for (int rep = 0; rep < 2; ++rep) {
    int colg = (wid + rep * 8) * 16 + (lane & 15);
    const bf16_t* brow = w1T + ((size_t)e * HMOE + colg) * DIM;
    v8f acc = {};
    for (int k0 = 0; k0 < DIM; k0 += 32)
      acc = wmma_bf16(frag_a(&Xs[rowA][0], k0, lh), frag_b(brow, k0, lh), acc);
    float bb = b1[e * HMOE + colg];
    for (int i = 0; i < 8; ++i) {
      int row = lh * 8 + i;
      Hsh[row][colg] = (bf16_t)silu(acc[i] + bb);
    }
  }
  __syncthreads();

  // GEMM2: Y = H @ W2 + b2; 64 N-tiles over 8 waves (8 each).
  for (int j = 0; j < 8; ++j) {
    int colg = (wid * 8 + j) * 16 + (lane & 15);
    const bf16_t* brow = w2T + ((size_t)e * DIM + colg) * HMOE;
    v8f acc = {};
    for (int k0 = 0; k0 < HMOE; k0 += 32)
      acc = wmma_bf16(frag_a(&Hsh[rowA][0], k0, lh), frag_b(brow, k0, lh), acc);
    float bb = b2[(size_t)e * DIM + colg];
    for (int i = 0; i < 8; ++i) {
      int row = lh * 8 + i;
      float v = (acc[i] + bb) * cwv[row];
      atomicAdd(&out[(size_t)toks[row] * DIM + colg], v);
    }
  }
}

// ---------------------------------------------------------------------------

extern "C" void kernel_launch(void* const* d_in, const int* in_sizes, int n_in,
                              void* d_out, int out_size, void* d_ws, size_t ws_size,
                              hipStream_t stream) {
  const float* x        = (const float*)d_in[0];
  const float* router_w = (const float*)d_in[1];
  const float* router_b = (const float*)d_in[2];
  const float* w1       = (const float*)d_in[3];
  const float* b1       = (const float*)d_in[4];
  const float* w2       = (const float*)d_in[5];
  const float* b2       = (const float*)d_in[6];
  const float* sw1      = (const float*)d_in[7];
  const float* sb1      = (const float*)d_in[8];
  const float* sw2      = (const float*)d_in[9];
  const float* sb2      = (const float*)d_in[10];
  float* out = (float*)d_out;

  // Workspace layout (~106 MB total).
  char* ws = (char*)d_ws;
  size_t off = 0;
  auto alloc = [&](size_t bytes) -> void* {
    void* p = ws + off;
    off += (bytes + 255) & ~(size_t)255;
    return p;
  };
  bf16_t* xbf   = (bf16_t*)alloc((size_t)T_TOK * DIM * sizeof(bf16_t));
  bf16_t* w1T   = (bf16_t*)alloc((size_t)NEXP * HMOE * DIM * sizeof(bf16_t));
  bf16_t* w2T   = (bf16_t*)alloc((size_t)NEXP * DIM * HMOE * sizeof(bf16_t));
  bf16_t* sw1T  = (bf16_t*)alloc((size_t)NSH * HSH * DIM * sizeof(bf16_t));
  bf16_t* sw2T  = (bf16_t*)alloc((size_t)NSH * DIM * HSH * sizeof(bf16_t));
  bf16_t* Hs    = (bf16_t*)alloc((size_t)NSH * T_TOK * HSH * sizeof(bf16_t));
  float*  cwbuf = (float*)alloc((size_t)T_TOK * NEXP * sizeof(float));
  int*    lists = (int*)alloc((size_t)NEXP * T_TOK * sizeof(int));
  int*    counts= (int*)alloc((size_t)NEXP * sizeof(int));
  (void)ws_size; (void)in_sizes; (void)n_in; (void)out_size;

  // 1) Precision/layout conversion (single streaming pass over all weights).
  cvt_bf16_kernel<<<512, 256, 0, stream>>>(x, xbf, (long)T_TOK * DIM);
  transpose_cvt_kernel<<<4096, 256, 0, stream>>>(w1,  w1T,  NEXP, DIM,  HMOE);
  transpose_cvt_kernel<<<4096, 256, 0, stream>>>(w2,  w2T,  NEXP, HMOE, DIM);
  transpose_cvt_kernel<<<2048, 256, 0, stream>>>(sw1, sw1T, NSH,  DIM,  HSH);
  transpose_cvt_kernel<<<2048, 256, 0, stream>>>(sw2, sw2T, NSH,  HSH,  DIM);

  // 2) Router + expert token lists.
  hipMemsetAsync(counts, 0, NEXP * sizeof(int), stream);
  router_kernel<<<T_TOK, 64, 0, stream>>>(x, router_w, router_b,
                                          cwbuf, counts, lists);

  // 3) Shared experts (dense write of out), then 4) routed atomics on top.
  shared_h_kernel<<<dim3(T_TOK / 16, HSH / 128, NSH), 256, 0, stream>>>(
      xbf, sw1T, sb1, Hs);
  shared_out_kernel<<<dim3(T_TOK / 16, DIM / 128), 256, 0, stream>>>(
      Hs, sw2T, sb2, out);
  routed_kernel<<<dim3(T_TOK / 16, NEXP), 256, 0, stream>>>(
      xbf, w1T, w2T, b1, b2, cwbuf, counts, lists, out);
}
